// ViTBlock_48687749267828
// MI455X (gfx1250) — matrix operand order repaired and checked by reading, not compiled, not run on hardware
//
#include <hip/hip_runtime.h>
#include <math.h>

// ---------------------------------------------------------------------------
// Types for CDNA5 WMMA (gfx1250, wave32)
// ---------------------------------------------------------------------------
typedef __bf16 bf16;
typedef __attribute__((ext_vector_type(8)))  bf16     bf16x8;
typedef __attribute__((ext_vector_type(16))) bf16     bf16x16;
typedef __attribute__((ext_vector_type(8)))  float    v8f;
typedef __attribute__((ext_vector_type(4)))  uint32_t u32x4;
typedef __attribute__((ext_vector_type(8)))  int      i32x8;
typedef __attribute__((ext_vector_type(4)))  int      i32x4;

union Frag { bf16x16 v; bf16x8 h[2]; };

#define LN_EPS 1e-5f

__device__ __forceinline__ bf16 f2bf(float f) { return (bf16)f; }
__device__ __forceinline__ float bf2f(bf16 b) { return (float)b; }

__device__ __forceinline__ float wave_sum(float v) {
    #pragma unroll
    for (int m = 16; m >= 1; m >>= 1) v += __shfl_xor(v, m, 32);
    return v;
}
__device__ __forceinline__ float wave_max(float v) {
    #pragma unroll
    for (int m = 16; m >= 1; m >>= 1) v = fmaxf(v, __shfl_xor(v, m, 32));
    return v;
}

// ---------------------------------------------------------------------------
// Problem constants
// ---------------------------------------------------------------------------
#define BATCH 16
#define IN_C  384
#define OUT_C 512
#define HH    32
#define WW    32
#define NPIX  1024            // H*W
#define BN_PIX 16384          // BATCH*NPIX
#define HEADS 8
#define HD    64
#define MID   512
#define HID   2048
#define QSCALE 0.125f         // 64^-0.5

// ---------------------------------------------------------------------------
// Tensor Data Mover (CDNA5) support: 2D tile global->LDS with HW padding.
// This toolchain exposes the 6-arg builtin form:
//   (u32x4 g0, i32x8 g1, i32x4 g2, i32x4 g3, i32x8 extra, i32 cpol)
// Guarded; falls back to the register-staged double-buffer path if absent.
// ---------------------------------------------------------------------------
#if defined(__has_builtin)
#  if __has_builtin(__builtin_amdgcn_tensor_load_to_lds) && \
      __has_builtin(__builtin_amdgcn_s_wait_tensorcnt)
#    define USE_TDM 1
#  endif
#endif

#ifdef USE_TDM
// Loads a (tile_h x 32) bf16 tile whose rows are row_elems apart in global
// memory into LDS at lds_byte_addr, with 16B of LDS padding inserted after
// every 64B row (pad_interval=3 -> 16 DWORDs, pad_amount=3 -> 4 DWORDs),
// i.e. LDS row stride = 80B = GLDT bf16 elements.
__device__ __forceinline__ void tdm_load_tile32(
    const bf16* gbase, uint32_t lds_byte_addr, int row_elems, int tile_h)
{
    uint64_t ga = (uint64_t)(uintptr_t)gbase;
    union { u32x4 v; uint32_t w[4]; } g0;
    g0.w[0] = 1u;                                     // count=1 (valid), user mode
    g0.w[1] = lds_byte_addr;                          // LDS dest (bytes)
    g0.w[2] = (uint32_t)ga;                           // global_addr[31:0]
    g0.w[3] = (uint32_t)((ga >> 32) & 0x01FFFFFFu)    // global_addr[56:32]
            | (2u << 30);                             // type=2 ("image")
    union { i32x8 v; uint64_t q[4]; } g1;
    const uint64_t dim0    = 32u;                     // tensor_dim0 (>= tile_w, OOB-safe)
    const uint64_t dim1    = (uint32_t)tile_h;        // tensor_dim1
    const uint64_t stride0 = (uint32_t)row_elems;     // tensor_dim0_stride (elems)
    g1.q[0] = (1ull << 16)                            // data_size = 1 -> 2 bytes
            | (1ull << 20)                            // pad_enable
            | (3ull << 22)                            // pad_interval: 16 DWORDs
            | (3ull << 25)                            // pad_amount:    4 DWORDs
            | (dim0 << 48);                           // tensor_dim0[15:0]
    g1.q[1] = (dim0 >> 16)                            // tensor_dim0[31:16]
            | (dim1 << 16)                            // tensor_dim1[31:0]
            | (32ull << 48);                          // tile_dim0 = 32
    g1.q[2] = (uint64_t)(uint16_t)tile_h              // tile_dim1
            | (stride0 << 32);                        // tensor_dim0_stride[31:0]
    g1.q[3] = (stride0 >> 32);                        // stride[47:32]; dim1_stride=0
    const i32x4 z4 = {0, 0, 0, 0};                    // 2D tensor: groups 2/3 unused
    const i32x8 z8 = {0, 0, 0, 0, 0, 0, 0, 0};        // extra descriptor words unused
    __builtin_amdgcn_tensor_load_to_lds(g0.v, g1.v, z4, z4, z8, 0);
}
#endif

// ---------------------------------------------------------------------------
// f32 -> bf16 weight convert
// ---------------------------------------------------------------------------
__global__ void cvt_bf16_kernel(const float* __restrict__ s, bf16* __restrict__ d, int n) {
    int i = blockIdx.x * 256 + threadIdx.x;
    if (i < n) d[i] = f2bf(s[i]);
}

// ---------------------------------------------------------------------------
// LN1 over channels of x[B,384,H,W]; emits xbf (raw) and xn (normalized),
// both bf16 [BN_PIX][384] row-major (pixel-major) for GEMM A operands.
// ---------------------------------------------------------------------------
__global__ __launch_bounds__(256) void ln1_pack_kernel(
    const float* __restrict__ x, const float* __restrict__ w, const float* __restrict__ bgain,
    bf16* __restrict__ xbf, bf16* __restrict__ xn)
{
    int p = blockIdx.x * 8 + (threadIdx.x >> 5);
    if (p >= BN_PIX) return;
    int l = threadIdx.x & 31;
    int b = p >> 10, n = p & 1023;
    float v[12];
    float sum = 0.f, sq = 0.f;
    #pragma unroll
    for (int i = 0; i < 12; ++i) {
        int c = l + 32 * i;
        float t = x[((size_t)(b * IN_C + c)) * NPIX + n];
        v[i] = t; sum += t; sq += t * t;
    }
    sum = wave_sum(sum); sq = wave_sum(sq);
    float mean = sum * (1.f / IN_C);
    float var  = sq * (1.f / IN_C) - mean * mean;
    float rs   = rsqrtf(fmaxf(var, 0.f) + LN_EPS);
    #pragma unroll
    for (int i = 0; i < 12; ++i) {
        int c = l + 32 * i;
        xbf[(size_t)p * IN_C + c] = f2bf(v[i]);
        xn [(size_t)p * IN_C + c] = f2bf((v[i] - mean) * rs * w[c] + bgain[c]);
    }
}

// ---------------------------------------------------------------------------
// LN2 over channels of f32 stream [BN_PIX][512] -> bf16 [BN_PIX][512]
// ---------------------------------------------------------------------------
__global__ __launch_bounds__(256) void ln2_kernel(
    const float* __restrict__ xpre, const float* __restrict__ w, const float* __restrict__ bgain,
    bf16* __restrict__ xn2)
{
    int p = blockIdx.x * 8 + (threadIdx.x >> 5);
    if (p >= BN_PIX) return;
    int l = threadIdx.x & 31;
    float v[16];
    float sum = 0.f, sq = 0.f;
    #pragma unroll
    for (int i = 0; i < 16; ++i) {
        int c = l + 32 * i;
        float t = xpre[(size_t)p * OUT_C + c];
        v[i] = t; sum += t; sq += t * t;
    }
    sum = wave_sum(sum); sq = wave_sum(sq);
    float mean = sum * (1.f / OUT_C);
    float var  = sq * (1.f / OUT_C) - mean * mean;
    float rs   = rsqrtf(fmaxf(var, 0.f) + LN_EPS);
    #pragma unroll
    for (int i = 0; i < 16; ++i) {
        int c = l + 32 * i;
        xn2[(size_t)p * OUT_C + c] = f2bf((v[i] - mean) * rs * w[c] + bgain[c]);
    }
}

// ---------------------------------------------------------------------------
// Generic bf16 WMMA GEMM:  D[M][N] = A[M][K] x W[N][K]^T  (+ epilogues)
//   Block tile 64x128, 8 waves; double-buffered LDS K-pipeline.
//   TDM path: wave 0 issues tensor_load_to_lds for the next K-tiles while all
//   waves run WMMA on the current buffer (TENSORcnt + barrier sync).
//   Fallback: register-staged global prefetch + ds_store into alternate buf.
// Epilogue modes:
//   0: outF = acc + bias
//   1: outF += (acc + bias) * gamma
//   2: outB = bf16(gelu_exact(acc + bias))
//   3: d_out NCHW-transposed: outF[((b*512+col)*1024)+n] = resid + (acc+bias)*gamma
// ---------------------------------------------------------------------------
#define GBM 64
#define GBN 128
#define GBK 32
#define GLDT 40   // padded LDS K-stride (bf16 elems) = 64B data + 16B pad

__global__ __launch_bounds__(256) void gemm_bf16_kernel(
    const bf16* __restrict__ A, int lda,
    const bf16* __restrict__ W, int ldw,
    const float* __restrict__ bias,
    const float* __restrict__ gamma,
    const float* __restrict__ resid,
    float* __restrict__ outF, bf16* __restrict__ outB, int ldd,
    int M, int N, int K, int mode)
{
    __shared__ __align__(16) bf16 As[2][GBM * GLDT];
    __shared__ __align__(16) bf16 Ws[2][GBN * GLDT];

    const int tid = threadIdx.x;
    const int l = tid & 31;
    const int w = tid >> 5;
    const int bm = blockIdx.y * GBM;
    const int bn = blockIdx.x * GBN;

    v8f acc[4] = {};
    const int ka = (l < 16) ? 0 : 8;    // A fragment K sub-offset
    const int ob = (l < 16) ? 0 : 8;    // B fragment K sub-offset
    const int nsteps = K / GBK;
    int buf = 0;

#ifdef USE_TDM
    if (w == 0) {
        tdm_load_tile32(&A[(size_t)bm * lda], (uint32_t)(uintptr_t)&As[0][0], lda, GBM);
        tdm_load_tile32(&W[(size_t)bn * ldw], (uint32_t)(uintptr_t)&Ws[0][0], ldw, GBN);
    }
    for (int i = 0; i < nsteps; ++i) {
        if (w == 0) __builtin_amdgcn_s_wait_tensorcnt(0);   // buf's DMA complete
        __syncthreads();                                    // publish buf; prev compute done
        if (w == 0 && i + 1 < nsteps) {
            const int kk = (i + 1) * GBK;
            tdm_load_tile32(&A[(size_t)bm * lda + kk],
                            (uint32_t)(uintptr_t)&As[buf ^ 1][0], lda, GBM);
            tdm_load_tile32(&W[(size_t)bn * ldw + kk],
                            (uint32_t)(uintptr_t)&Ws[buf ^ 1][0], ldw, GBN);
        }
        // ---- compute current buffer ----
        Frag bfr;
        const int wr = w * 16 + (l & 15);
        bfr.h[0] = *(const bf16x8*)&Ws[buf][wr * GLDT + ob];
        bfr.h[1] = *(const bf16x8*)&Ws[buf][wr * GLDT + ob + 16];
        #pragma unroll
        for (int rt = 0; rt < 4; ++rt) {
            Frag afr;
            const int ar = rt * 16 + (l & 15);
            afr.h[0] = *(const bf16x8*)&As[buf][ar * GLDT + ka];
            afr.h[1] = *(const bf16x8*)&As[buf][ar * GLDT + ka + 16];
            acc[rt] = __builtin_amdgcn_wmma_f32_16x16x32_bf16(
                false, afr.v, false, bfr.v, (short)0, acc[rt], false, false);
        }
        buf ^= 1;
    }
#else
    const int arw = tid >> 2, arc = (tid & 3) * 8;     // A tile: 64 x 32, 16B/thread
    const int wrw = tid >> 1, wrc = (tid & 1) * 16;    // W tile: 128 x 32, 32B/thread
    bf16x8 ra, rw0, rw1;
    ra  = *(const bf16x8*)&A[(size_t)(bm + arw) * lda + arc];
    rw0 = *(const bf16x8*)&W[(size_t)(bn + wrw) * ldw + wrc];
    rw1 = *(const bf16x8*)&W[(size_t)(bn + wrw) * ldw + wrc + 8];
    *(bf16x8*)&As[0][arw * GLDT + arc]     = ra;
    *(bf16x8*)&Ws[0][wrw * GLDT + wrc]     = rw0;
    *(bf16x8*)&Ws[0][wrw * GLDT + wrc + 8] = rw1;
    for (int i = 0; i < nsteps; ++i) {
        const bool more = (i + 1 < nsteps);
        if (more) {   // global prefetch of next K-tile into registers
            const int kk = (i + 1) * GBK;
            ra  = *(const bf16x8*)&A[(size_t)(bm + arw) * lda + kk + arc];
            rw0 = *(const bf16x8*)&W[(size_t)(bn + wrw) * ldw + kk + wrc];
            rw1 = *(const bf16x8*)&W[(size_t)(bn + wrw) * ldw + kk + wrc + 8];
        }
        __syncthreads();   // current buf visible; prev compute done (alt buf reusable)
        Frag bfr;
        const int wr = w * 16 + (l & 15);
        bfr.h[0] = *(const bf16x8*)&Ws[buf][wr * GLDT + ob];
        bfr.h[1] = *(const bf16x8*)&Ws[buf][wr * GLDT + ob + 16];
        #pragma unroll
        for (int rt = 0; rt < 4; ++rt) {
            Frag afr;
            const int ar = rt * 16 + (l & 15);
            afr.h[0] = *(const bf16x8*)&As[buf][ar * GLDT + ka];
            afr.h[1] = *(const bf16x8*)&As[buf][ar * GLDT + ka + 16];
            acc[rt] = __builtin_amdgcn_wmma_f32_16x16x32_bf16(
                false, afr.v, false, bfr.v, (short)0, acc[rt], false, false);
        }
        if (more) {   // stage prefetched tile into the alternate buffer
            *(bf16x8*)&As[buf ^ 1][arw * GLDT + arc]     = ra;
            *(bf16x8*)&Ws[buf ^ 1][wrw * GLDT + wrc]     = rw0;
            *(bf16x8*)&Ws[buf ^ 1][wrw * GLDT + wrc + 8] = rw1;
        }
        buf ^= 1;
    }
#endif

    const int colg = bn + w * 16 + (l & 15);
    const float bb = bias  ? bias[colg]  : 0.f;
    const float gg = gamma ? gamma[colg] : 1.f;
    const int rbase = (l >> 4) << 3;   // 0 or 8

    #pragma unroll
    for (int rt = 0; rt < 4; ++rt) {
        #pragma unroll
        for (int r = 0; r < 8; ++r) {
            int rowg = bm + rt * 16 + r + rbase;
            float v = acc[rt][r] + bb;
            if (mode == 0) {
                outF[(size_t)rowg * ldd + colg] = v;
            } else if (mode == 1) {
                outF[(size_t)rowg * ldd + colg] += v * gg;
            } else if (mode == 2) {
                float g = 0.5f * v * (1.f + erff(v * 0.70710678118654752f));
                outB[(size_t)rowg * ldd + colg] = f2bf(g);
            } else { // mode 3: final residual, NCHW transposed store
                int b = rowg >> 10, n = rowg & 1023;
                float res = resid[(size_t)rowg * OUT_C + colg];
                outF[(((size_t)b * OUT_C + colg) << 10) + n] = res + v * gg;
            }
        }
    }
}

// ---------------------------------------------------------------------------
// Pool branch: depthwise 3x3 (zero pad) + channel LN over HD=64, per head.
// One wave per (b,h,n); lanes handle d = lane and lane+32.
// ---------------------------------------------------------------------------
__global__ __launch_bounds__(256) void pool_kernel(
    const float* __restrict__ in, int ldin, int coff,
    const float* __restrict__ w3,       // [64][9]
    const float* __restrict__ lnw, const float* __restrict__ lnb,
    float scale,
    bf16* __restrict__ rowOut, bf16* __restrict__ tOut)
{
    int gw = blockIdx.x * 8 + (threadIdx.x >> 5);   // = bh*1024 + n
    if (gw >= BATCH * HEADS * NPIX) return;
    int l = threadIdx.x & 31;
    int bh = gw >> 10, n = gw & 1023;
    int b = bh >> 3, h = bh & 7;
    int y = n >> 5, xx0 = n & 31;
    int cbase = coff + h * HD;

    float vals[2];
    #pragma unroll
    for (int j = 0; j < 2; ++j) {
        int d = l + 32 * j;
        float s = 0.f;
        #pragma unroll
        for (int dy = -1; dy <= 1; ++dy) {
            int yy = y + dy;
            if ((unsigned)yy >= (unsigned)HH) continue;
            #pragma unroll
            for (int dx = -1; dx <= 1; ++dx) {
                int xc = xx0 + dx;
                if ((unsigned)xc >= (unsigned)WW) continue;
                int p = b * NPIX + yy * WW + xc;
                s += w3[d * 9 + (dy + 1) * 3 + (dx + 1)] *
                     in[(size_t)p * ldin + cbase + d];
            }
        }
        vals[j] = s;
    }
    float sum = wave_sum(vals[0] + vals[1]);
    float sq  = wave_sum(vals[0] * vals[0] + vals[1] * vals[1]);
    float mean = sum * (1.f / HD);
    float var  = sq * (1.f / HD) - mean * mean;
    float rs   = rsqrtf(fmaxf(var, 0.f) + LN_EPS);
    #pragma unroll
    for (int j = 0; j < 2; ++j) {
        int d = l + 32 * j;
        float o = ((vals[j] - mean) * rs * lnw[d] + lnb[d]) * scale;
        if (rowOut) rowOut[(size_t)gw * HD + d] = f2bf(o);
        if (tOut)   tOut[((size_t)(bh * HD + d)) * NPIX + n] = f2bf(o);
    }
}

// ---------------------------------------------------------------------------
// Attention: one block per (b,h, 16-query-row slab).
//   S = q_slab @ k^T (WMMA, rel-pos bias in epilogue), softmax in LDS f32,
//   P -> bf16 LDS (padded), O = P @ v (WMMA) + pooled-q residual.
// Dynamic LDS: 16*1024*4 + 16*1032*2 = 98560 B
// ---------------------------------------------------------------------------
#define PLD 1032

__global__ __launch_bounds__(256) void attn_kernel(
    const bf16* __restrict__ qp, const bf16* __restrict__ kp, const bf16* __restrict__ vT,
    const float* __restrict__ rel_table, bf16* __restrict__ attnA)
{
    extern __shared__ char smem[];
    float* S = (float*)smem;                       // [16][1024]
    bf16*  P = (bf16*)(smem + 16 * 1024 * 4);      // [16][PLD]

    const int blk = blockIdx.x;
    const int rb = blk & 63;          // query row-block (16 rows)
    const int bh = blk >> 6;          // 0..127
    const int b = bh >> 3, h = bh & 7;
    const int tid = threadIdx.x;
    const int l = tid & 31;
    const int w = tid >> 5;
    const int ka = (l < 16) ? 0 : 8;
    const int ob = (l < 16) ? 0 : 8;
    const size_t qbase = (size_t)bh * NPIX * HD;

    // Hoisted A fragments (q rows) for both K-steps of K=64
    Frag af0, af1;
    {
        const int qrow = rb * 16 + (l & 15);
        const bf16* qr = qp + qbase + (size_t)qrow * HD;
        af0.h[0] = *(const bf16x8*)&qr[ka];
        af0.h[1] = *(const bf16x8*)&qr[ka + 16];
        af1.h[0] = *(const bf16x8*)&qr[32 + ka];
        af1.h[1] = *(const bf16x8*)&qr[32 + ka + 16];
    }

    // ---- Step 1: S tiles (64 column tiles split over 8 waves) ----
    for (int ct = w; ct < 64; ct += 8) {
        const int mcol = ct * 16 + (l & 15);
        const bf16* kr = kp + qbase + (size_t)mcol * HD;
        Frag bf0, bf1;
        bf0.h[0] = *(const bf16x8*)&kr[ob];
        bf0.h[1] = *(const bf16x8*)&kr[ob + 16];
        bf1.h[0] = *(const bf16x8*)&kr[32 + ob];
        bf1.h[1] = *(const bf16x8*)&kr[32 + ob + 16];
        v8f acc = {};
        acc = __builtin_amdgcn_wmma_f32_16x16x32_bf16(false, af0.v, false, bf0.v, (short)0, acc, false, false);
        acc = __builtin_amdgcn_wmma_f32_16x16x32_bf16(false, af1.v, false, bf1.v, (short)0, acc, false, false);

        const int y2 = mcol >> 5, x2 = mcol & 31;
        const int rbase = (l >> 4) << 3;
        #pragma unroll
        for (int r = 0; r < 8; ++r) {
            int row_l = r + rbase;
            int nrow = rb * 16 + row_l;
            int y1 = nrow >> 5, x1 = nrow & 31;
            int idx = (y1 - y2 + HH - 1) * (2 * WW - 1) + (x1 - x2 + WW - 1);
            S[row_l * 1024 + mcol] = acc[r] + rel_table[idx * HEADS + h];
        }
    }
    __syncthreads();

    // ---- Step 2: softmax, 2 rows per wave ----
    #pragma unroll
    for (int rr = 0; rr < 2; ++rr) {
        int row = w * 2 + rr;
        float mx = -3.0e38f;
        for (int c = l; c < 1024; c += 32) mx = fmaxf(mx, S[row * 1024 + c]);
        mx = wave_max(mx);
        float sum = 0.f;
        for (int c = l; c < 1024; c += 32) {
            float e = expf(S[row * 1024 + c] - mx);
            S[row * 1024 + c] = e;
            sum += e;
        }
        sum = wave_sum(sum);
        float inv = 1.f / sum;
        for (int c = l; c < 1024; c += 32)
            P[row * PLD + c] = f2bf(S[row * 1024 + c] * inv);
    }
    __syncthreads();

    // ---- Step 3: O = P @ v (waves 0..3, one 16x16 d-tile each) ----
    if (w < 4) {
        v8f acc = {};
        const int dcol = w * 16 + (l & 15);
        const bf16* vr = vT + ((size_t)(bh * HD + dcol)) * NPIX;
        const int arow = (l & 15);
        for (int kk = 0; kk < 1024; kk += 32) {
            Frag afr, bfr;
            afr.h[0] = *(const bf16x8*)&P[arow * PLD + kk + ka];
            afr.h[1] = *(const bf16x8*)&P[arow * PLD + kk + ka + 16];
            bfr.h[0] = *(const bf16x8*)&vr[kk + ob];
            bfr.h[1] = *(const bf16x8*)&vr[kk + ob + 16];
            acc = __builtin_amdgcn_wmma_f32_16x16x32_bf16(false, afr.v, false, bfr.v, (short)0, acc, false, false);
        }
        const int rbase = (l >> 4) << 3;
        #pragma unroll
        for (int r = 0; r < 8; ++r) {
            int ng = rb * 16 + r + rbase;
            float v = acc[r] + bf2f(qp[qbase + (size_t)ng * HD + dcol]);
            attnA[((size_t)(b * NPIX + ng)) * MID + h * HD + dcol] = f2bf(v);
        }
    }
}

// ---------------------------------------------------------------------------
// Host-side launch
// ---------------------------------------------------------------------------
extern "C" void kernel_launch(void* const* d_in, const int* in_sizes, int n_in,
                              void* d_out, int out_size, void* d_ws, size_t ws_size,
                              hipStream_t stream) {
    const float* x           = (const float*)d_in[0];
    const float* norm1_w     = (const float*)d_in[1];
    const float* norm1_b     = (const float*)d_in[2];
    const float* q_w         = (const float*)d_in[3];
    const float* q_b         = (const float*)d_in[4];
    const float* kv_w        = (const float*)d_in[5];
    const float* kv_b        = (const float*)d_in[6];
    const float* pool_q_w    = (const float*)d_in[7];
    const float* pool_k_w    = (const float*)d_in[8];
    const float* pool_v_w    = (const float*)d_in[9];
    const float* pnq_w       = (const float*)d_in[10];
    const float* pnq_b       = (const float*)d_in[11];
    const float* pnk_w       = (const float*)d_in[12];
    const float* pnk_b       = (const float*)d_in[13];
    const float* pnv_w       = (const float*)d_in[14];
    const float* pnv_b       = (const float*)d_in[15];
    const float* rel_table   = (const float*)d_in[16];
    const float* attn_proj_w = (const float*)d_in[17];
    const float* attn_proj_b = (const float*)d_in[18];
    const float* attn_gamma  = (const float*)d_in[19];
    const float* pool_skip_w = (const float*)d_in[20];
    const float* pool_skip_b = (const float*)d_in[21];
    const float* norm2_w     = (const float*)d_in[22];
    const float* norm2_b     = (const float*)d_in[23];
    const float* fc1_w       = (const float*)d_in[24];
    const float* fc1_b       = (const float*)d_in[25];
    const float* fc2_w       = (const float*)d_in[26];
    const float* fc2_b       = (const float*)d_in[27];
    const float* mlp_gamma   = (const float*)d_in[28];
    (void)in_sizes; (void)n_in; (void)out_size; (void)ws_size;

    // ---- workspace carve-out (region reuse across stage lifetimes) ----
    char* ws = (char*)d_ws;
    size_t off = 0;
    auto alloc = [&](size_t bytes) -> void* {
        off = (off + 255) & ~(size_t)255;
        void* p = ws + off;
        off += bytes;
        return p;
    };
    bf16* wq    = (bf16*)alloc((size_t)MID  * IN_C * 2);
    bf16* wkv   = (bf16*)alloc((size_t)2*MID* IN_C * 2);
    bf16* wproj = (bf16*)alloc((size_t)OUT_C* MID  * 2);
    bf16* wskip = (bf16*)alloc((size_t)OUT_C* IN_C * 2);
    bf16* wfc1  = (bf16*)alloc((size_t)HID  * OUT_C* 2);
    bf16* wfc2  = (bf16*)alloc((size_t)OUT_C* HID  * 2);
    bf16* xbf   = (bf16*)alloc((size_t)BN_PIX * IN_C * 2);
    bf16* xnrm  = (bf16*)alloc((size_t)BN_PIX * OUT_C * 2);          // LN1 out (ld 384) / LN2 out (ld 512)
    void* reg2  = alloc((size_t)BN_PIX * MID * 4);                   // qconv f32 | attnA bf16
    void* reg3  = alloc((size_t)BN_PIX * 2 * MID * 4);               // kvconv f32 | hmid bf16
    bf16* qp    = (bf16*)alloc((size_t)BATCH*HEADS*NPIX * HD * 2);
    bf16* kp    = (bf16*)alloc((size_t)BATCH*HEADS*NPIX * HD * 2);
    bf16* vT    = (bf16*)alloc((size_t)BATCH*HEADS*HD * NPIX * 2);
    float* xpre = (float*)alloc((size_t)BN_PIX * OUT_C * 4);

    // ---- 1. weights -> bf16 ----
    auto cvt = [&](const float* s, bf16* d, int n) {
        cvt_bf16_kernel<<<(n + 255) / 256, 256, 0, stream>>>(s, d, n);
    };
    cvt(q_w,         wq,    MID * IN_C);
    cvt(kv_w,        wkv,   2 * MID * IN_C);
    cvt(attn_proj_w, wproj, OUT_C * MID);
    cvt(pool_skip_w, wskip, OUT_C * IN_C);
    cvt(fc1_w,       wfc1,  HID * OUT_C);
    cvt(fc2_w,       wfc2,  OUT_C * HID);

    // ---- 2. LN1 + bf16 pack ----
    ln1_pack_kernel<<<BN_PIX / 8, 256, 0, stream>>>(x, norm1_w, norm1_b, xbf, xnrm);

    // ---- 3. q / kv projections (f32 out for pooling) ----
    gemm_bf16_kernel<<<dim3(MID / GBN, BN_PIX / GBM), 256, 0, stream>>>(
        xnrm, IN_C, wq, IN_C, q_b, nullptr, nullptr,
        (float*)reg2, nullptr, MID, BN_PIX, MID, IN_C, 0);
    gemm_bf16_kernel<<<dim3(2 * MID / GBN, BN_PIX / GBM), 256, 0, stream>>>(
        xnrm, IN_C, wkv, IN_C, kv_b, nullptr, nullptr,
        (float*)reg3, nullptr, 2 * MID, BN_PIX, 2 * MID, IN_C, 0);

    // ---- 4. pool branches (dwconv3x3 + channel LN) ----
    const int poolBlocks = BATCH * HEADS * NPIX / 8;
    pool_kernel<<<poolBlocks, 256, 0, stream>>>(
        (const float*)reg2, MID, 0, pool_q_w, pnq_w, pnq_b, QSCALE, qp, nullptr);
    pool_kernel<<<poolBlocks, 256, 0, stream>>>(
        (const float*)reg3, 2 * MID, 0, pool_k_w, pnk_w, pnk_b, 1.0f, kp, nullptr);
    pool_kernel<<<poolBlocks, 256, 0, stream>>>(
        (const float*)reg3, 2 * MID, MID, pool_v_w, pnv_w, pnv_b, 1.0f, nullptr, vT);

    // ---- 5. attention (S -> softmax -> P@V + q residual) ----
    const size_t attnShm = (size_t)16 * 1024 * 4 + (size_t)16 * PLD * 2;  // 98560 B
    attn_kernel<<<BATCH * HEADS * (NPIX / 16), 256, attnShm, stream>>>(
        qp, kp, vT, rel_table, (bf16*)reg2);

    // ---- 6. skip GEMM (f32 stream), then proj GEMM accumulates with gamma ----
    gemm_bf16_kernel<<<dim3(OUT_C / GBN, BN_PIX / GBM), 256, 0, stream>>>(
        xbf, IN_C, wskip, IN_C, pool_skip_b, nullptr, nullptr,
        xpre, nullptr, OUT_C, BN_PIX, OUT_C, IN_C, 0);
    gemm_bf16_kernel<<<dim3(OUT_C / GBN, BN_PIX / GBM), 256, 0, stream>>>(
        (const bf16*)reg2, MID, wproj, MID, attn_proj_b, attn_gamma, nullptr,
        xpre, nullptr, OUT_C, BN_PIX, OUT_C, MID, 1);

    // ---- 7. MLP: LN2 -> fc1(gelu) -> fc2 (+residual, transposed NCHW out) ----
    ln2_kernel<<<BN_PIX / 8, 256, 0, stream>>>(xpre, norm2_w, norm2_b, xnrm);
    gemm_bf16_kernel<<<dim3(HID / GBN, BN_PIX / GBM), 256, 0, stream>>>(
        xnrm, OUT_C, wfc1, OUT_C, fc1_b, nullptr, nullptr,
        nullptr, (bf16*)reg3, HID, BN_PIX, HID, OUT_C, 2);
    gemm_bf16_kernel<<<dim3(OUT_C / GBN, BN_PIX / GBM), 256, 0, stream>>>(
        (const bf16*)reg3, HID, wfc2, HID, fc2_b, mlp_gamma, xpre,
        (float*)d_out, nullptr, OUT_C, BN_PIX, OUT_C, HID, 3);
}
